// Encoder_49855980372583
// MI455X (gfx1250) — matrix operand order repaired and checked by reading, not compiled, not run on hardware
//
#include <hip/hip_runtime.h>

#define IN_DIM 128
#define UNITS  256
#define GDIM   768      // 3*UNITS
#define BATCH  128
#define TSTEPS 1024

typedef __attribute__((ext_vector_type(16))) __bf16 v16bf;
typedef __attribute__((ext_vector_type(8)))  float  v8f;

union V16U { v16bf v; unsigned short u[16]; uint4 q[2]; };
union V8U  { v8f   v; float f[8]; };

__device__ __forceinline__ unsigned short f32_to_bf16(float x) {
  unsigned int u = __float_as_uint(x);
  unsigned int r = u + 0x7FFFu + ((u >> 16) & 1u);   // round-to-nearest-even
  return (unsigned short)(r >> 16);
}

__device__ __forceinline__ float sigmoidf_(float v) {
  return 1.0f / (1.0f + __expf(-v));
}

// ---------------------------------------------------------------------------
// Kernel 1: W,U -> bf16 *transposed* (col-major: [n][k]); zero sync counters.
// Makes every WMMA B-fragment two contiguous 16B loads.
// ---------------------------------------------------------------------------
__global__ void gru_prep_kernel(const float* __restrict__ W, const float* __restrict__ U,
                                unsigned short* __restrict__ WbfT, unsigned short* __restrict__ UbfT,
                                unsigned int* __restrict__ counters) {
  const int stride = gridDim.x * blockDim.x;
  const int idx = blockIdx.x * blockDim.x + threadIdx.x;
  for (int i = idx; i < GDIM * IN_DIM; i += stride) {   // WbfT[n*128 + k]
    const int n = i / IN_DIM, k = i % IN_DIM;
    WbfT[i] = f32_to_bf16(W[(long)k * GDIM + n]);
  }
  for (int i = idx; i < GDIM * UNITS; i += stride) {    // UbfT[n*256 + k]
    const int n = i / UNITS, k = i % UNITS;
    UbfT[i] = f32_to_bf16(U[(long)k * GDIM + n]);
  }
  if (idx < 8) counters[idx] = 0u;
}

// ---------------------------------------------------------------------------
// Kernel 2: x (f32) -> xbf (bf16), vectorized float4 -> 2x packed u32
// ---------------------------------------------------------------------------
__global__ __launch_bounds__(256) void gru_xcvt_kernel(const float* __restrict__ x,
                                                       unsigned int* __restrict__ xbfu) {
  const size_t i = (size_t)blockIdx.x * blockDim.x + threadIdx.x;   // 4 elems each
  const float4 v = ((const float4*)x)[i];
  const unsigned int lo = (unsigned)f32_to_bf16(v.x) | ((unsigned)f32_to_bf16(v.y) << 16);
  const unsigned int hi = (unsigned)f32_to_bf16(v.z) | ((unsigned)f32_to_bf16(v.w) << 16);
  uint2 p; p.x = lo; p.y = hi;
  ((uint2*)xbfu)[i] = p;
}

// ---------------------------------------------------------------------------
// Kernel 3: x_proj = x @ W + b_i   (131072 x 128) @ (128 x 768)
// One wave -> 16x64 tile; A fragments preloaded (4 chunks), all operand loads
// are b128.
// ---------------------------------------------------------------------------
__global__ __launch_bounds__(256) void gru_xproj_kernel(
    const unsigned short* __restrict__ xbf, const unsigned short* __restrict__ WbfT,
    const float* __restrict__ bvec, float* __restrict__ xp) {
  const int lane  = threadIdx.x & 31;
  const int row   = lane & 15;
  const int khalf = lane >> 4;
  const long wid  = (long)blockIdx.x * 8 + (threadIdx.x >> 5);
  const int  NB   = GDIM / 64;                 // 12
  const long mt   = wid / NB;
  const int  nb   = (int)(wid % NB);

  // A fragments for all 4 K-chunks (b128 x 2 each)
  V16U a[4];
  const unsigned short* xr = xbf + (size_t)(mt * 16 + row) * IN_DIM + khalf * 8;
#pragma unroll
  for (int kc = 0; kc < 4; ++kc) {
    a[kc].q[0] = *(const uint4*)(xr + kc * 32);
    a[kc].q[1] = *(const uint4*)(xr + kc * 32 + 16);
  }

#pragma unroll
  for (int s = 0; s < 4; ++s) {
    const int col = nb * 64 + s * 16 + row;
    const unsigned short* wp = WbfT + (size_t)col * IN_DIM + khalf * 8;
    v8f acc = v8f{};
#pragma unroll
    for (int kc = 0; kc < 4; ++kc) {
      V16U bm;
      bm.q[0] = *(const uint4*)(wp + kc * 32);
      bm.q[1] = *(const uint4*)(wp + kc * 32 + 16);
      acc = __builtin_amdgcn_wmma_f32_16x16x32_bf16(
          false, a[kc].v, false, bm.v, (short)0, acc, false, false);
    }
    const float bias = bvec[col];
    V8U c; c.v = acc;
#pragma unroll
    for (int i = 0; i < 8; ++i) {
      const int M = i + (khalf << 3);
      xp[(mt * 16 + M) * (long)GDIM + col] = c.f[i] + bias;
    }
  }
}

// ---------------------------------------------------------------------------
// Kernel 4: recurrence. 8 batch-groups x 16 WGs; 16 batches x 48 cols/gate
// per WG per step. U fragments live in VGPRs for all 1024 steps; h staged via
// async global->LDS (ASYNCcnt, GV addressing); A from LDS b128; per-group
// atomic sync.
// ---------------------------------------------------------------------------
__global__ __launch_bounds__(288) void gru_recur_kernel(
    const float* __restrict__ xp, const unsigned short* __restrict__ UbfT,
    const float* __restrict__ bvec, const float* __restrict__ hid,
    unsigned int* __restrict__ counters, float* __restrict__ out) {
  __shared__ float          hf32[16][UNITS];   // h_{t-1} exact (async-staged)
  __shared__ unsigned short hbf[16][UNITS];    // h_{t-1} bf16 (A operand)
  __shared__ float          recbuf[9][16][16]; // rec tiles for gate fusion

  const int tid   = threadIdx.x;
  const int lane  = tid & 31;
  const int w     = tid >> 5;                  // wave 0..8
  const int row   = lane & 15;
  const int khalf = lane >> 4;
  const int group = blockIdx.x >> 4;           // 0..7
  const int wgIn  = blockIdx.x & 15;           // 0..15
  const int gate  = w / 3;
  const int sub   = w % 3;
  const int col0  = gate * 256 + wgIn * 48 + sub * 16;
  const float* br = bvec + GDIM;

  // ---- preload this wave's U fragments for all 8 K-chunks (time-invariant)
  V16U ub[8];
  {
    const unsigned short* up = UbfT + (size_t)(col0 + row) * UNITS + khalf * 8;
#pragma unroll
    for (int kc = 0; kc < 8; ++kc) {
      ub[kc].q[0] = *(const uint4*)(up + kc * 32);
      ub[kc].q[1] = *(const uint4*)(up + kc * 32 + 16);
    }
  }

  const unsigned hf32_base = (unsigned)(size_t)&hf32[0][0];

  for (int t = 0; t < TSTEPS; ++t) {
    // ---- wait for all 16 WGs of this group to have published h_{t-1}
    if (t > 0 && lane == 0) {
      while (__hip_atomic_load(&counters[group], __ATOMIC_ACQUIRE,
                               __HIP_MEMORY_SCOPE_AGENT) < 16u * (unsigned)t)
        __builtin_amdgcn_s_sleep(1);
    }
    __syncthreads();

    // ---- async-stage h_{t-1} (16 x 256 f32 = 16KB) into LDS (GV addressing)
    {
      const unsigned long long srcbase =
          (t == 0) ? (unsigned long long)(const void*)hid
                   : (unsigned long long)(const void*)(out + (size_t)(t - 1) * UNITS);
      const unsigned rowpitch = (t == 0) ? (UNITS * 4u) : (TSTEPS * UNITS * 4u);
      for (int i = tid; i < 1024; i += 288) {             // 1024 x b128
        const int r = i >> 6, c4 = i & 63;
        const unsigned long long gaddr =
            srcbase + (unsigned long long)((unsigned)(group * 16 + r) * rowpitch + c4 * 16u);
        const unsigned ldsa = hf32_base + (unsigned)r * (UNITS * 4u) + c4 * 16u;
        asm volatile("global_load_async_to_lds_b128 %0, %1, off"
                     :: "v"(ldsa), "v"(gaddr) : "memory");
      }
      asm volatile("s_wait_asynccnt 0x0" ::: "memory");
    }
    __syncthreads();

    // ---- convert staged h to bf16 for the A operand (float4 -> 4x bf16)
    for (int i = tid; i < 1024; i += 288) {
      const int r = i >> 6, c4 = (i & 63) << 2;
      const float4 v = *(const float4*)&hf32[r][c4];
      hbf[r][c4 + 0] = f32_to_bf16(v.x);
      hbf[r][c4 + 1] = f32_to_bf16(v.y);
      hbf[r][c4 + 2] = f32_to_bf16(v.z);
      hbf[r][c4 + 3] = f32_to_bf16(v.w);
    }
    __syncthreads();

    // ---- rec tile: (16 x 256) @ (256 x 16), A via ds_load_b128
    v8f acc = v8f{};
#pragma unroll
    for (int kc = 0; kc < 8; ++kc) {
      V16U a;
      const unsigned short* hp8 = &hbf[row][kc * 32 + khalf * 8];
      a.q[0] = *(const uint4*)hp8;
      a.q[1] = *(const uint4*)(hp8 + 16);
      acc = __builtin_amdgcn_wmma_f32_16x16x32_bf16(
          false, a.v, false, ub[kc].v, (short)0, acc, false, false);
    }
    {
      V8U c; c.v = acc;
#pragma unroll
      for (int i = 0; i < 8; ++i)
        recbuf[w][i + (khalf << 3)][row] = c.f[i];
    }
    __syncthreads();

    // ---- gate fusion + h_t store (waves 0..2)
    if (w < 3) {
      const int cgl = wgIn * 48 + w * 16 + row;
#pragma unroll
      for (int i = 0; i < 8; ++i) {
        const int M = i + (khalf << 3);
        const int bidx = group * 16 + M;
        const long m = (long)bidx * TSTEPS + t;
        const float xz = xp[m * GDIM + cgl];
        const float xr = xp[m * GDIM + 256 + cgl];
        const float xh = xp[m * GDIM + 512 + cgl];
        const float rz = recbuf[0 + w][M][row] + br[cgl];
        const float rr = recbuf[3 + w][M][row] + br[256 + cgl];
        const float rh = recbuf[6 + w][M][row] + br[512 + cgl];
        const float hp = hf32[M][cgl];
        const float z  = sigmoidf_(xz + rz);
        const float r  = sigmoidf_(xr + rr);
        const float hh = tanhf(xh + r * rh);
        out[((long)bidx * TSTEPS + t) * UNITS + cgl] = z * hp + (1.0f - z) * hh;
      }
    }
    __syncthreads();

    // ---- publish h_t
    if (tid == 0) {
      __threadfence();
      __hip_atomic_fetch_add(&counters[group], 1u, __ATOMIC_RELEASE,
                             __HIP_MEMORY_SCOPE_AGENT);
    }
  }
}

// ---------------------------------------------------------------------------
// Kernel 5: final state = h_{T-1}
// ---------------------------------------------------------------------------
__global__ void gru_state_kernel(float* __restrict__ out) {
  const int idx = blockIdx.x * blockDim.x + threadIdx.x;
  if (idx < BATCH * UNITS) {
    const int b = idx / UNITS, c = idx % UNITS;
    out[(long)BATCH * TSTEPS * UNITS + idx] =
        out[((long)b * TSTEPS + (TSTEPS - 1)) * UNITS + c];
  }
}

extern "C" void kernel_launch(void* const* d_in, const int* in_sizes, int n_in,
                              void* d_out, int out_size, void* d_ws, size_t ws_size,
                              hipStream_t stream) {
  const float* x   = (const float*)d_in[0];
  const float* hid = (const float*)d_in[1];
  const float* W   = (const float*)d_in[2];
  const float* U   = (const float*)d_in[3];
  const float* b   = (const float*)d_in[4];
  float* out = (float*)d_out;

  char* ws = (char*)d_ws;
  unsigned short* WbfT     = (unsigned short*)(ws);                 // 192 KB
  unsigned short* UbfT     = (unsigned short*)(ws + 196608);        // 384 KB
  unsigned int*   counters = (unsigned int*)(ws + 589824);          // 32 B
  unsigned short* xbf      = (unsigned short*)(ws + 1048576);       // 32 MB
  float*          xp       = (float*)(ws + 34603008);               // 384 MB

  hipLaunchKernelGGL(gru_prep_kernel, dim3(1152), dim3(256), 0, stream,
                     W, U, WbfT, UbfT, counters);
  hipLaunchKernelGGL(gru_xcvt_kernel, dim3(16384), dim3(256), 0, stream,
                     x, (unsigned int*)xbf);
  hipLaunchKernelGGL(gru_xproj_kernel, dim3(12288), dim3(256), 0, stream,
                     xbf, WbfT, b, xp);
  hipLaunchKernelGGL(gru_recur_kernel, dim3(128), dim3(288), 0, stream,
                     xp, UbfT, b, hid, counters, out);
  hipLaunchKernelGGL(gru_state_kernel, dim3((BATCH * UNITS + 255) / 256), dim3(256),
                     0, stream, out);
}
